// SelfAttention2_13073880449268
// MI455X (gfx1250) — matrix operand order repaired and checked by reading, not compiled, not run on hardware
//
#include <hip/hip_runtime.h>
#include <hip/hip_bf16.h>
#include <stdint.h>

typedef __attribute__((ext_vector_type(16))) _Float16 v16h;
typedef __attribute__((ext_vector_type(8)))  _Float16 v8h;
typedef __attribute__((ext_vector_type(8)))  float    v8f;
typedef __attribute__((ext_vector_type(4)))  int      v4i;

#define B_  8
#define C_  256
#define K_  32
#define N_  4096

__device__ __forceinline__ v16h cat8(v8h lo, v8h hi) {
  return __builtin_shufflevector(lo, hi, 0,1,2,3,4,5,6,7,8,9,10,11,12,13,14,15);
}

__device__ __forceinline__ v8f wmma_f16(v16h a, v16h b, v8f c) {
  // D = A(16x32 f16) * B(32x16 f16) + C(16x16 f32)
  return __builtin_amdgcn_wmma_f32_16x16x32_f16(false, a, false, b,
                                                (short)0, c, false, false);
}

// Async global->LDS copy path (gfx1250 ASYNCcnt-tracked), with sync fallback.
#if defined(__HIP_DEVICE_COMPILE__) && \
    __has_builtin(__builtin_amdgcn_global_load_async_to_lds_b128) && \
    __has_builtin(__builtin_amdgcn_s_wait_asynccnt)
#define USE_ASYNC 1
#else
#define USE_ASYNC 0
#endif

#if USE_ASYNC
typedef __attribute__((address_space(1))) v4i* as1_v4i_p;  // global
typedef __attribute__((address_space(3))) v4i* as3_v4i_p;  // LDS
__device__ __forceinline__ void async_copy16(const void* g, void* lds) {
  // Generic LDS pointers carry the LDS offset in the low 32 bits.
  __builtin_amdgcn_global_load_async_to_lds_b128(
      (as1_v4i_p)(uintptr_t)g,
      (as3_v4i_p)(unsigned)(uintptr_t)lds,
      0, 0);
}
#endif

// ---------------------------------------------------------------------------
// Kernel 1: theta/phi/g projections (1x1 conv 256->32) via WMMA.
// Outputs: thetaH [b][n][32] f16, phiH [b][n][32] f16, gTH [b][32][n] f16.
// ---------------------------------------------------------------------------
__global__ __launch_bounds__(256) void nl_proj_kernel(
    const float* __restrict__ x,
    const float* __restrict__ w_theta, const float* __restrict__ b_theta,
    const float* __restrict__ w_phi,   const float* __restrict__ b_phi,
    const float* __restrict__ w_g,     const float* __restrict__ b_g,
    _Float16* __restrict__ thetaH, _Float16* __restrict__ phiH,
    _Float16* __restrict__ gTH)
{
  const int b    = blockIdx.y;
  const int n0   = blockIdx.x * 64;
  const int tid  = threadIdx.x;
  const int wave = tid >> 5, lane = tid & 31;
  const int ln   = lane & 15, half = lane >> 4;

  __shared__ _Float16 xT[64 * 264];       // [n][c] f16, stride 264 (pad: no bank conflicts)
  __shared__ _Float16 wH[3 * 32 * 264];   // [(p*32+k)][c] f16, stride 264

  // Stage x tile transposed -> xT[n][c] (f32 -> f16)
  for (int idx = tid; idx < 256 * 64; idx += 256) {
    int c = idx >> 6, n = idx & 63;
    xT[n * 264 + c] = (_Float16)x[((size_t)(b * C_ + c)) * N_ + n0 + n];
  }
  // Stage all three weight matrices (f32 -> f16)
  for (int idx = tid; idx < 3 * 32 * 256; idx += 256) {
    int p = idx >> 13, rem = idx & 8191;
    int k = rem >> 8, c = rem & 255;
    const float* wp = (p == 0) ? w_theta : (p == 1) ? w_phi : w_g;
    wH[(p * 32 + k) * 264 + c] = (_Float16)wp[k * 256 + c];
  }
  __syncthreads();

  // 24 output tiles (3 proj x 4 n-tiles x 2 k-tiles); 8 waves x 3 tiles each.
  for (int t = 0; t < 3; ++t) {
    int tile = wave * 3 + t;          // uniform per wave
    int p  = tile >> 3, sub = tile & 7;
    int nt = sub >> 1,  kt  = sub & 1;
    const float* bp = (p == 0) ? b_theta : (p == 1) ? b_phi : b_g;
    float bv = bp[kt * 16 + ln];
    v8f acc = {bv, bv, bv, bv, bv, bv, bv, bv};
    #pragma unroll
    for (int cb = 0; cb < 8; ++cb) {
      int c0 = cb * 32;
      // A tile: rows = positions, K = channels (A-layout per ISA 7.12.2)
      const _Float16* ar = &xT[(nt * 16 + ln) * 264 + c0 + half * 8];
      v16h a = cat8(*(const v8h*)ar, *(const v8h*)(ar + 16));
      // B tile: rows = channels, cols = k outputs (B-layout: same K order as A)
      const _Float16* br = &wH[(p * 32 + kt * 16 + ln) * 264 + c0 + half * 8];
      v16h bm = cat8(*(const v8h*)br, *(const v8h*)(br + 16));
      acc = wmma_f16(a, bm, acc);
    }
    #pragma unroll
    for (int r = 0; r < 8; ++r) {
      int m = r + half * 8;                 // C-layout row
      int n = n0 + nt * 16 + m;
      int k = kt * 16 + ln;                 // C-layout column
      _Float16 h = (_Float16)acc[r];
      if (p == 0)      thetaH[((size_t)b * N_ + n) * K_ + k] = h;
      else if (p == 1) phiH[((size_t)b * N_ + n) * K_ + k]   = h;
      else             gTH[(size_t)b * K_ * N_ + (size_t)k * N_ + n] = h;
    }
  }
}

// ---------------------------------------------------------------------------
// Kernel 2: fused flash-attention. WG = 64 rows (4 waves x 16 rows), loop over
// all 4096 j in tiles of 32, double-buffered async global->LDS staging.
// Never materializes the 4096x4096 matrix.
// ---------------------------------------------------------------------------
__global__ __launch_bounds__(128) void nl_attn_kernel(
    const _Float16* __restrict__ thetaH, const _Float16* __restrict__ phiH,
    const _Float16* __restrict__ gTH, float* __restrict__ O)
{
  const int b    = blockIdx.y;
  const int tid  = threadIdx.x;
  const int wave = tid >> 5, lane = tid & 31;
  const int ln   = lane & 15, half = lane >> 4;
  const int rowBase = blockIdx.x * 64 + wave * 16;
  const int NT = N_ / 32;

  __shared__ _Float16 sPhi[2][32 * 40];   // [j][k] tiles, stride 40 (pad)
  __shared__ _Float16 sGT [2][32 * 40];   // [k][j] tiles
  __shared__ _Float16 sP  [4 * 16 * 40];  // per-wave P round-trip (D->A layout)
  _Float16* sPw = &sP[wave * 16 * 40];

  // Q tile (theta rows), loaded once per wave in A-layout
  const _Float16* qr = thetaH + ((size_t)b * N_ + rowBase + ln) * K_ + half * 8;
  v16h aQ = cat8(*(const v8h*)qr, *(const v8h*)(qr + 16));

  v8f acc0 = {}; v8f acc1 = {};           // O accumulator: k=0..15, k=16..31
  float mrun[8], lrun[8];
  #pragma unroll
  for (int r = 0; r < 8; ++r) { mrun[r] = -1e30f; lrun[r] = 0.f; }

  const int ldRow = tid >> 2;
  const int ldCol = (tid & 3) * 8;
  const _Float16* phiBase = phiH + ((size_t)b * N_) * K_;
  const _Float16* gBase   = gTH  + (size_t)b * K_ * N_;

  // Stage one 32-j tile (each wave: 2 x 512B async copies -> ASYNCcnt += 2)
  auto stage = [&](int buf, int j0) {
#if USE_ASYNC
    async_copy16(&phiBase[(size_t)(j0 + ldRow) * K_ + ldCol],
                 &sPhi[buf][ldRow * 40 + ldCol]);
    async_copy16(&gBase[(size_t)ldRow * N_ + j0 + ldCol],
                 &sGT[buf][ldRow * 40 + ldCol]);
#else
    *(v8h*)&sPhi[buf][ldRow * 40 + ldCol] =
        *(const v8h*)&phiBase[(size_t)(j0 + ldRow) * K_ + ldCol];
    *(v8h*)&sGT[buf][ldRow * 40 + ldCol] =
        *(const v8h*)&gBase[(size_t)ldRow * N_ + j0 + ldCol];
    if (j0 + 32 < N_) {
      __builtin_prefetch(&phiBase[(size_t)(j0 + 32 + ldRow) * K_ + ldCol], 0, 0);
      __builtin_prefetch(&gBase[(size_t)ldRow * N_ + j0 + 32 + ldCol], 0, 0);
    }
#endif
  };

  stage(0, 0);
  for (int jt = 0; jt < NT; ++jt) {
    const int cur = jt & 1;
    const int j0  = jt * 32;
    // Prefetch next tile into the other buffer (freed by last iter's barrier)
    if (jt + 1 < NT) stage(cur ^ 1, j0 + 32);
#if USE_ASYNC
    // Async loads retire in order per wave: <=2 outstanding => tile jt landed.
    if (jt + 1 < NT) __builtin_amdgcn_s_wait_asynccnt(2);
    else             __builtin_amdgcn_s_wait_asynccnt(0);
#endif
    __syncthreads();                       // tile `cur` visible to all waves

    // S = Q . phi^T : two 16x16 j-tiles, K=32
    const _Float16* p0r = &sPhi[cur][ln * 40 + half * 8];
    v16h bP0 = cat8(*(const v8h*)p0r, *(const v8h*)(p0r + 16));
    const _Float16* p1r = &sPhi[cur][(16 + ln) * 40 + half * 8];
    v16h bP1 = cat8(*(const v8h*)p1r, *(const v8h*)(p1r + 16));
    v8f zc = {};
    v8f s0 = wmma_f16(aQ, bP0, zc);
    v8f s1 = wmma_f16(aQ, bP1, zc);

    // Online softmax: row values live across 16 lanes of one half (C-layout)
    #pragma unroll
    for (int r = 0; r < 8; ++r) {
      float t = fmaxf(s0[r], s1[r]);
      t = fmaxf(t, __shfl_xor(t, 1));
      t = fmaxf(t, __shfl_xor(t, 2));
      t = fmaxf(t, __shfl_xor(t, 4));
      t = fmaxf(t, __shfl_xor(t, 8));
      float mn = fmaxf(mrun[r], t);
      float sc = __expf(mrun[r] - mn);
      float p0 = __expf(s0[r] - mn);
      float p1 = __expf(s1[r] - mn);
      float rs = p0 + p1;
      rs += __shfl_xor(rs, 1); rs += __shfl_xor(rs, 2);
      rs += __shfl_xor(rs, 4); rs += __shfl_xor(rs, 8);
      lrun[r] = lrun[r] * sc + rs;
      mrun[r] = mn;
      acc0[r] *= sc; acc1[r] *= sc;
      sPw[(r + half * 8) * 40 + ln]      = (_Float16)p0;  // D-layout -> LDS
      sPw[(r + half * 8) * 40 + 16 + ln] = (_Float16)p1;
    }
    // LDS is in-order per wave; wait for DS stores, block compiler reordering
    asm volatile("s_wait_dscnt 0" ::: "memory");

    // Read P back in A-layout; B = gT tiles; accumulate O += P * g^T
    const _Float16* pr = &sPw[ln * 40 + half * 8];
    v16h aP = cat8(*(const v8h*)pr, *(const v8h*)(pr + 16));
    const _Float16* g0 = &sGT[cur][ln * 40 + half * 8];
    v16h bG0 = cat8(*(const v8h*)g0, *(const v8h*)(g0 + 16));
    const _Float16* g1 = &sGT[cur][(16 + ln) * 40 + half * 8];
    v16h bG1 = cat8(*(const v8h*)g1, *(const v8h*)(g1 + 16));
    acc0 = wmma_f16(aP, bG0, acc0);
    acc1 = wmma_f16(aP, bG1, acc1);

    __syncthreads();   // all waves done reading `cur` before it is restaged
  }

  #pragma unroll
  for (int r = 0; r < 8; ++r) {
    float inv = 1.0f / lrun[r];
    int row = rowBase + r + half * 8;
    float* orow = O + ((size_t)b * N_ + row) * K_;
    orow[ln]      = acc0[r] * inv;
    orow[16 + ln] = acc1[r] * inv;
  }
}

// ---------------------------------------------------------------------------
// Kernel 3: out = x + sigma * (W_attn @ O + b_attn). 32 MACs per output.
// ---------------------------------------------------------------------------
__global__ __launch_bounds__(256) void nl_out_kernel(
    const float* __restrict__ x, const float* __restrict__ w_attn,
    const float* __restrict__ b_attn, const float* __restrict__ sigma,
    const float* __restrict__ O, float* __restrict__ out)
{
  const int b  = blockIdx.y;
  const int n0 = blockIdx.x * 64;
  const int c  = threadIdx.x;
  __shared__ float sO[64 * 32];
  for (int idx = c; idx < 64 * 32; idx += 256) {
    int row = idx >> 5, k = idx & 31;
    sO[idx] = O[((size_t)b * N_ + n0 + row) * K_ + k];
  }
  float wr[32];
  #pragma unroll
  for (int k = 0; k < 32; ++k) wr[k] = w_attn[c * 32 + k];
  const float bias = b_attn[c];
  const float sg   = sigma[0];
  __syncthreads();
  for (int n = 0; n < 64; ++n) {
    float acc = bias;
    #pragma unroll
    for (int k = 0; k < 32; ++k) acc += wr[k] * sO[n * 32 + k];  // LDS broadcast
    size_t idx = ((size_t)(b * C_ + c)) * N_ + n0 + n;
    out[idx] = x[idx] + sg * acc;
  }
}

// ---------------------------------------------------------------------------
extern "C" void kernel_launch(void* const* d_in, const int* in_sizes, int n_in,
                              void* d_out, int out_size, void* d_ws, size_t ws_size,
                              hipStream_t stream) {
  const float* x       = (const float*)d_in[0];
  const float* w_theta = (const float*)d_in[1];
  const float* b_theta = (const float*)d_in[2];
  const float* w_phi   = (const float*)d_in[3];
  const float* b_phi   = (const float*)d_in[4];
  const float* w_g     = (const float*)d_in[5];
  const float* b_g     = (const float*)d_in[6];
  const float* w_attn  = (const float*)d_in[7];
  const float* b_attn  = (const float*)d_in[8];
  const float* sigma   = (const float*)d_in[9];
  float* out = (float*)d_out;

  const size_t nk = (size_t)B_ * N_ * K_;      // 1M elements
  _Float16* thetaH = (_Float16*)d_ws;          // 2 MB
  _Float16* phiH   = thetaH + nk;              // 2 MB
  _Float16* gTH    = phiH + nk;                // 2 MB
  float*    O      = (float*)(gTH + nk);       // 4 MB  (total 10 MB of d_ws)

  dim3 grid(N_ / 64, B_);
  nl_proj_kernel<<<grid, 256, 0, stream>>>(x, w_theta, b_theta, w_phi, b_phi,
                                           w_g, b_g, thetaH, phiH, gTH);
  nl_attn_kernel<<<grid, 128, 0, stream>>>(thetaH, phiH, gTH, O);
  nl_out_kernel<<<grid, 256, 0, stream>>>(x, w_attn, b_attn, sigma, O, out);
}
